// PMTBlock_86500641341617
// MI455X (gfx1250) — matrix operand order repaired and checked by reading, not compiled, not run on hardware
//
#include <hip/hip_runtime.h>
#include <hip/hip_bf16.h>
#include <math.h>

#define NS 2048
#define ND 512
#define NCH 32
#define NFE 16
#define NHD 8

typedef __attribute__((ext_vector_type(16))) _Float16 v16h;
typedef __attribute__((ext_vector_type(8)))  _Float16 v8h;
typedef __attribute__((ext_vector_type(8)))  float    v8f;

// ---------------------------------------------------------------------------
// WMMA fragment helpers (wave32, v_wmma_f32_16x16x32_f16; contiguous k = 16*(lane/16) + e, same order in A and B)
// ---------------------------------------------------------------------------

// A matrix 16x32 f16, row-major source with leading dim lda.
// lane L: M = L%16 ; element e -> K = kb + e (16 contiguous f16)
// where kb = (L>=16 ? 16 : 0).
__device__ inline v16h load_a_frag(const _Float16* __restrict__ A, int lda, int m0, int k0) {
    int lane = threadIdx.x & 31;
    int m  = m0 + (lane & 15);
    int kb = k0 + ((lane >> 4) << 4);
    const _Float16* p = A + (size_t)m * lda + kb;
    v8h lo = *reinterpret_cast<const v8h*>(p);
    v8h hi = *reinterpret_cast<const v8h*>(p + 8);
    return __builtin_shufflevector(lo, hi, 0,1,2,3,4,5,6,7,8,9,10,11,12,13,14,15);
}

// B matrix 32x16 f16 from K-major (transposed) storage BT[N][K] with leading dim ldk.
// lane L: N = L%16 ; element e -> K = (L>=16 ? 16 : 0) + e  (16 contiguous f16).
__device__ inline v16h load_bt_frag(const _Float16* __restrict__ BT, int ldk, int n0, int k0) {
    int lane = threadIdx.x & 31;
    int n  = n0 + (lane & 15);
    int kb = k0 + ((lane >> 4) << 4);
    const _Float16* p = BT + (size_t)n * ldk + kb;
    v8h lo = *reinterpret_cast<const v8h*>(p);
    v8h hi = *reinterpret_cast<const v8h*>(p + 8);
    return __builtin_shufflevector(lo, hi, 0,1,2,3,4,5,6,7,8,9,10,11,12,13,14,15);
}

#define WMMA_F16(a, b, c) \
    __builtin_amdgcn_wmma_f32_16x16x32_f16(false, (a), false, (b), (short)0, (c), false, false)

// ---------------------------------------------------------------------------
// Kernel: l2 normalize features over the C (=32) axis. One block per point s.
// ---------------------------------------------------------------------------
__global__ __launch_bounds__(256) void l2norm_kernel(const float* __restrict__ in,
                                                     float* __restrict__ out) {
    int s = blockIdx.x;
    int t = threadIdx.x;
#pragma unroll
    for (int half = 0; half < 2; ++half) {
        int idx = t + half * 256;                       // idx = f*32 + c, wave = one f-group
        float v  = in[(size_t)s * ND + idx];
        float sq = v * v;
#pragma unroll
        for (int off = 16; off > 0; off >>= 1)
            sq += __shfl_xor(sq, off, 32);
        float nrm = sqrtf(sq);
        out[(size_t)s * ND + idx] = v / fmaxf(nrm, 1e-12f);
    }
}

// ---------------------------------------------------------------------------
// Kernel: attention probabilities. One block per query row s.
// A[h][s][j] = softmax_j( mask(d) ? -inf : mlp(d)_h ) as f16.
// ---------------------------------------------------------------------------
__global__ __launch_bounds__(256) void attn_kernel(const float* __restrict__ pos,  // [S,3]
                                                   const float* __restrict__ w1,   // [8]
                                                   const float* __restrict__ b1,   // [8]
                                                   const float* __restrict__ w2,   // [8,8]
                                                   const float* __restrict__ b2,   // [8]
                                                   _Float16* __restrict__ A) {     // [8][S][S]
    __shared__ float Sm[NHD * 256];
    __shared__ float Sl[NHD * 256];
    int s = blockIdx.x;
    int t = threadIdx.x;

    float px = pos[s * 3 + 0], py = pos[s * 3 + 1], pz = pos[s * 3 + 2];
    float W1[8], B1[8], B2[8], W2[64];
#pragma unroll
    for (int k = 0; k < 8; ++k) { W1[k] = w1[k]; B1[k] = b1[k]; B2[k] = b2[k]; }
#pragma unroll
    for (int k = 0; k < 64; ++k) W2[k] = w2[k];

    float m[8], l[8];
#pragma unroll
    for (int h = 0; h < 8; ++h) { m[h] = -1e30f; l[h] = 0.f; }

    // pass 1: online (max, sumexp) per head
    for (int j = t; j < NS; j += 256) {
        float dx = px - pos[j * 3 + 0];
        float dy = py - pos[j * 3 + 1];
        float dz = pz - pos[j * 3 + 2];
        float d  = sqrtf(dx * dx + dy * dy + dz * dz);
        if (d <= 2.5f) {
            float hid[8];
#pragma unroll
            for (int k = 0; k < 8; ++k) hid[k] = fmaxf(d * W1[k] + B1[k], 0.f);
#pragma unroll
            for (int h = 0; h < 8; ++h) {
                float logit = B2[h];
#pragma unroll
                for (int k = 0; k < 8; ++k) logit += hid[k] * W2[k * 8 + h];
                if (logit > m[h]) {
                    l[h] = l[h] * __expf(m[h] - logit) + 1.f;
                    m[h] = logit;
                } else {
                    l[h] += __expf(logit - m[h]);
                }
            }
        }
    }
#pragma unroll
    for (int h = 0; h < 8; ++h) { Sm[h * 256 + t] = m[h]; Sl[h * 256 + t] = l[h]; }
    __syncthreads();
    for (int st = 128; st >= 1; st >>= 1) {
        if (t < st) {
#pragma unroll
            for (int h = 0; h < 8; ++h) {
                float m1 = Sm[h * 256 + t],      l1 = Sl[h * 256 + t];
                float m2 = Sm[h * 256 + t + st], l2 = Sl[h * 256 + t + st];
                float M = fmaxf(m1, m2);
                Sm[h * 256 + t] = M;
                Sl[h * 256 + t] = l1 * __expf(m1 - M) + l2 * __expf(m2 - M);
            }
        }
        __syncthreads();
    }
    float Mh[8], Lh[8];
#pragma unroll
    for (int h = 0; h < 8; ++h) { Mh[h] = Sm[h * 256]; Lh[h] = Sl[h * 256]; }

    // pass 2: write normalized probs
    for (int j = t; j < NS; j += 256) {
        float dx = px - pos[j * 3 + 0];
        float dy = py - pos[j * 3 + 1];
        float dz = pz - pos[j * 3 + 2];
        float d  = sqrtf(dx * dx + dy * dy + dz * dz);
        bool ok = (d <= 2.5f);
        float hid[8];
#pragma unroll
        for (int k = 0; k < 8; ++k) hid[k] = fmaxf(d * W1[k] + B1[k], 0.f);
#pragma unroll
        for (int h = 0; h < 8; ++h) {
            float p = 0.f;
            if (ok) {
                float logit = B2[h];
#pragma unroll
                for (int k = 0; k < 8; ++k) logit += hid[k] * W2[k * 8 + h];
                p = __expf(logit - Mh[h]) / Lh[h];
            }
            A[((size_t)h * NS + s) * NS + j] = (_Float16)p;
        }
    }
}

// ---------------------------------------------------------------------------
// Kernel: value projection + proxy. One block per point s.
// V[h][s][o*32+c] = sum_f (x[s]@ww + wb)[f, c*8+h] * prox[h,f,o]   (f16 out)
// ---------------------------------------------------------------------------
__global__ __launch_bounds__(256) void vproj_kernel(const float* __restrict__ X,    // [S,512]
                                                    const float* __restrict__ ww,   // [32,256]
                                                    const float* __restrict__ wb,   // [256]
                                                    const float* __restrict__ prox, // [8,16,16]
                                                    _Float16* __restrict__ V) {     // [8][S][512]
    __shared__ float xs[ND];
    __shared__ float tmp[NFE * 256];
    int s = blockIdx.x;
    int t = threadIdx.x;
    xs[t]       = X[(size_t)s * ND + t];
    xs[t + 256] = X[(size_t)s * ND + t + 256];
    __syncthreads();
#pragma unroll
    for (int f = 0; f < NFE; ++f) {
        float acc = wb[t];
#pragma unroll
        for (int c = 0; c < NCH; ++c) acc += xs[f * 32 + c] * ww[c * 256 + t];
        tmp[f * 256 + t] = acc;
    }
    __syncthreads();
    int c = t >> 3, h = t & 7;
#pragma unroll
    for (int o = 0; o < 16; ++o) {
        float acc = 0.f;
#pragma unroll
        for (int f = 0; f < NFE; ++f)
            acc += tmp[f * 256 + c * 8 + h] * prox[h * 256 + f * 16 + o];
        V[((size_t)h * NS + s) * ND + o * 32 + c] = (_Float16)acc;
    }
}

// ---------------------------------------------------------------------------
// Kernel: tiled f16 transpose per head plane: out[c][r] = in[r][c]
// grid = (cols/32, rows/32, planes), block = 256
// ---------------------------------------------------------------------------
__global__ __launch_bounds__(256) void transpose_f16_kernel(const _Float16* __restrict__ in,
                                                            _Float16* __restrict__ out,
                                                            int rows, int cols) {
    __shared__ _Float16 tile[32][40];
    size_t plane = (size_t)rows * cols;
    const _Float16* inp  = in  + (size_t)blockIdx.z * plane;
    _Float16*       outp = out + (size_t)blockIdx.z * plane;
    int r0 = blockIdx.y * 32, c0 = blockIdx.x * 32;
    int tc = threadIdx.x & 31;
    int tr = threadIdx.x >> 5;  // 0..7
#pragma unroll
    for (int i = 0; i < 4; ++i) {
        int r = tr + i * 8;
        tile[r][tc] = inp[(size_t)(r0 + r) * cols + (c0 + tc)];
    }
    __syncthreads();
#pragma unroll
    for (int i = 0; i < 4; ++i) {
        int r = tr + i * 8;
        outp[(size_t)(c0 + r) * rows + (r0 + tc)] = tile[tc][r];
    }
}

// ---------------------------------------------------------------------------
// Kernel: f32 -> f16 convert + transpose: out[c][r] = (f16)in[r][c]
// grid = (cols/32, rows/32), block = 256
// ---------------------------------------------------------------------------
__global__ __launch_bounds__(256) void cvtT_f16_kernel(const float* __restrict__ in,
                                                       _Float16* __restrict__ out,
                                                       int rows, int cols) {
    __shared__ float tile[32][33];
    int r0 = blockIdx.y * 32, c0 = blockIdx.x * 32;
    int tc = threadIdx.x & 31;
    int tr = threadIdx.x >> 5;
#pragma unroll
    for (int i = 0; i < 4; ++i) {
        int r = tr + i * 8;
        tile[r][tc] = in[(size_t)(r0 + r) * cols + (c0 + tc)];
    }
    __syncthreads();
#pragma unroll
    for (int i = 0; i < 4; ++i) {
        int r = tr + i * 8;
        out[(size_t)(c0 + r) * rows + (r0 + tc)] = (_Float16)tile[tc][r];
    }
}

// ---------------------------------------------------------------------------
// Kernel: batched WMMA GEMM.  C[h] = A[h] (MxK f16, row-major) * B[h]
// (KxN f16, given K-major as BT[N][K]) [+ bias].
// Each wave computes a 64x64 register tile: 16 f32 accumulators,
// 16 v_wmma_f32_16x16x32_f16 per K-step, all operand loads are b128.
// ---------------------------------------------------------------------------
__global__ __launch_bounds__(256) void wmma_gemm_kernel(const _Float16* __restrict__ A,
                                                        const _Float16* __restrict__ BT,
                                                        float* __restrict__ C,
                                                        const float* __restrict__ bias,
                                                        int M, int N, int K, int nHeads,
                                                        size_t strideA, size_t strideB,
                                                        size_t strideC) {
    int wid  = threadIdx.x >> 5;
    int tile = blockIdx.x * 8 + wid;
    int nt64 = N >> 6;
    int mt64 = M >> 6;
    int perHead = nt64 * mt64;
    int h = tile / perHead;
    if (h >= nHeads) return;
    int r  = tile - h * perHead;
    int mt = r / nt64;
    int nt = r - mt * nt64;
    int m0 = mt << 6, n0 = nt << 6;

    const _Float16* Ah = A  + (size_t)h * strideA;
    const _Float16* Bh = BT + (size_t)h * strideB;
    float*          Ch = C  + (size_t)h * strideC;

    int lane = threadIdx.x & 31;
    v8f zero = {0.f, 0.f, 0.f, 0.f, 0.f, 0.f, 0.f, 0.f};
    v8f acc[4][4];
#pragma unroll
    for (int mi = 0; mi < 4; ++mi)
#pragma unroll
        for (int ni = 0; ni < 4; ++ni) acc[mi][ni] = zero;

    for (int k0 = 0; k0 < K; k0 += 32) {
        // prefetch one 128B cacheline (2 K-steps) ahead, every other iteration
        if ((k0 & 63) == 0 && (k0 + 64) < K) {
            int kpA = k0 + 64 + ((lane >> 4) << 3);
            int kpB = k0 + 64 + ((lane >> 4) << 4);
#pragma unroll
            for (int mi = 0; mi < 4; ++mi)
                __builtin_prefetch(Ah + (size_t)(m0 + mi * 16 + (lane & 15)) * K + kpA, 0, 0);
#pragma unroll
            for (int ni = 0; ni < 4; ++ni)
                __builtin_prefetch(Bh + (size_t)(n0 + ni * 16 + (lane & 15)) * K + kpB, 0, 0);
        }
        v16h a0 = load_a_frag(Ah, K, m0,      k0);
        v16h a1 = load_a_frag(Ah, K, m0 + 16, k0);
        v16h a2 = load_a_frag(Ah, K, m0 + 32, k0);
        v16h a3 = load_a_frag(Ah, K, m0 + 48, k0);
#pragma unroll
        for (int ni = 0; ni < 4; ++ni) {
            v16h b = load_bt_frag(Bh, K, n0 + ni * 16, k0);
            acc[0][ni] = WMMA_F16(a0, b, acc[0][ni]);
            acc[1][ni] = WMMA_F16(a1, b, acc[1][ni]);
            acc[2][ni] = WMMA_F16(a2, b, acc[2][ni]);
            acc[3][ni] = WMMA_F16(a3, b, acc[3][ni]);
        }
    }

    int moff = (lane >> 4) << 3;
    int nl   = lane & 15;
#pragma unroll
    for (int ni = 0; ni < 4; ++ni) {
        int nn = n0 + ni * 16 + nl;
        float bv = bias ? bias[nn] : 0.f;
#pragma unroll
        for (int mi = 0; mi < 4; ++mi)
#pragma unroll
            for (int g = 0; g < 8; ++g)
                Ch[(size_t)(m0 + mi * 16 + g + moff) * N + nn] = acc[mi][ni][g] + bv;
    }
}

// ---------------------------------------------------------------------------
// Kernel: epilogue. One block per point s.
// z[f,e] = ob[e] + sum_{c,h} Y[h][s][f*32+c] * ow[(c*8+h)*32+e]
// X[s]   = leakyrelu(groupnorm_f(z) * g + b)
// ---------------------------------------------------------------------------
__global__ __launch_bounds__(256) void epilogue_kernel(const float* __restrict__ Y,  // [8][S][512]
                                                       const float* __restrict__ ow, // [256,32]
                                                       const float* __restrict__ ob, // [32]
                                                       const float* __restrict__ gg, // [32]
                                                       const float* __restrict__ gb, // [32]
                                                       float* __restrict__ X) {      // [S,512]
    __shared__ float y[NHD * ND];
    __shared__ float z[ND];
    __shared__ float mu[NCH], inv[NCH];
    int s = blockIdx.x;
    int t = threadIdx.x;
    for (int i = t; i < NHD * ND; i += 256) {
        int h = i >> 9, n = i & 511;
        y[i] = Y[((size_t)h * NS + s) * ND + n];
    }
    __syncthreads();
    for (int idx = t; idx < ND; idx += 256) {
        int f = idx >> 5, e = idx & 31;
        float acc = ob[e];
#pragma unroll
        for (int c = 0; c < NCH; ++c)
#pragma unroll
            for (int h = 0; h < NHD; ++h)
                acc += y[h * ND + f * 32 + c] * ow[(c * 8 + h) * 32 + e];
        z[idx] = acc;
    }
    __syncthreads();
    if (t < NCH) {
        float m = 0.f;
#pragma unroll
        for (int f = 0; f < NFE; ++f) m += z[f * 32 + t];
        m *= (1.f / NFE);
        float v = 0.f;
#pragma unroll
        for (int f = 0; f < NFE; ++f) { float d = z[f * 32 + t] - m; v += d * d; }
        v *= (1.f / NFE);
        mu[t]  = m;
        inv[t] = rsqrtf(v + 1e-5f);
    }
    __syncthreads();
    for (int idx = t; idx < ND; idx += 256) {
        int e = idx & 31;
        float val = (z[idx] - mu[e]) * inv[e] * gg[e] + gb[e];
        val = (val >= 0.f) ? val : 0.01f * val;
        X[(size_t)s * ND + idx] = val;
    }
}

// ---------------------------------------------------------------------------
// Kernel: f32 -> f16 convert (no transpose)
// ---------------------------------------------------------------------------
__global__ __launch_bounds__(256) void cvt_f16_kernel(const float* __restrict__ in,
                                                      _Float16* __restrict__ out, int n) {
    int i = blockIdx.x * 256 + threadIdx.x;
    if (i < n) out[i] = (_Float16)in[i];
}

// ---------------------------------------------------------------------------
// Host launcher
// ---------------------------------------------------------------------------
extern "C" void kernel_launch(void* const* d_in, const int* in_sizes, int n_in,
                              void* d_out, int out_size, void* d_ws, size_t ws_size,
                              hipStream_t stream) {
    (void)in_sizes; (void)n_in; (void)out_size; (void)ws_size;

    const float* src_pos  = (const float*)d_in[0];
    const float* trg_pos  = (const float*)d_in[1];
    const float* src_feat = (const float*)d_in[2];
    const float* trg_feat = (const float*)d_in[3];
    const float* wsrc_w   = (const float*)d_in[4];
    const float* wsrc_b   = (const float*)d_in[5];
    const float* wtrg_w   = (const float*)d_in[6];
    const float* wtrg_b   = (const float*)d_in[7];
    const float* osrc_w   = (const float*)d_in[8];
    const float* osrc_b   = (const float*)d_in[9];
    const float* otrg_w   = (const float*)d_in[10];
    const float* otrg_b   = (const float*)d_in[11];
    const float* proxy    = (const float*)d_in[12];
    const float* a1w1     = (const float*)d_in[13];
    const float* a1b1     = (const float*)d_in[14];
    const float* a1w2     = (const float*)d_in[15];
    const float* a1b2     = (const float*)d_in[16];
    const float* a2w1     = (const float*)d_in[17];
    const float* a2b1     = (const float*)d_in[18];
    const float* a2w2     = (const float*)d_in[19];
    const float* a2b2     = (const float*)d_in[20];
    const float* gn_g     = (const float*)d_in[21];
    const float* gn_b     = (const float*)d_in[22];
    const float* out_w    = (const float*)d_in[23];
    const float* out_b    = (const float*)d_in[24];

    char* ws = (char*)d_ws;
    _Float16* Abuf = (_Float16*)(ws);                                 // 67,108,864 B
    _Float16* Vbuf = (_Float16*)(ws + 67108864ull);                   // 16,777,216 B
    _Float16* VT   = (_Float16*)(ws + 83886080ull);                   // 16,777,216 B
    float*    Ybuf = (float*)   (ws + 100663296ull);                  // 33,554,432 B
    float*    Xsrc = (float*)   (ws + 134217728ull);                  //  4,194,304 B
    float*    Xtrg = (float*)   (ws + 138412032ull);                  //  4,194,304 B
    _Float16* Xh   = (_Float16*)(ws + 142606336ull);                  //  2,097,152 B
    _Float16* WT   = (_Float16*)(ws + 144703488ull);                  //    524,288 B

    const size_t SA = (size_t)NS * NS;       // per-head A stride
    const size_t SV = (size_t)NS * ND;       // per-head V / VT / Y stride

    // initial l2 normalization
    l2norm_kernel<<<NS, 256, 0, stream>>>(src_feat, Xsrc);
    l2norm_kernel<<<NS, 256, 0, stream>>>(trg_feat, Xtrg);

    const int gemm_attn_blocks = (NHD * (NS / 64) * (ND / 64)) / 8;   // 256
    const int gemm_out_blocks  = ((NS / 64) * (ND / 64)) / 8;         // 32
    const dim3 trV(ND / 32, NS / 32, NHD);                            // V transpose grid
    const dim3 trW(ND / 32, ND / 32, 1);                              // out_w transpose grid

    for (int i = 0; i < 2; ++i) {
        // ---- src branch ----
        attn_kernel<<<NS, 256, 0, stream>>>(src_pos, a1w1 + i * 8, a1b1 + i * 8,
                                            a1w2 + i * 64, a1b2 + i * 8, Abuf);
        vproj_kernel<<<NS, 256, 0, stream>>>(Xsrc, wsrc_w + i * 8192, wsrc_b + i * 256,
                                             proxy + i * 2048, Vbuf);
        transpose_f16_kernel<<<trV, 256, 0, stream>>>(Vbuf, VT, NS, ND);
        wmma_gemm_kernel<<<gemm_attn_blocks, 256, 0, stream>>>(
            Abuf, VT, Ybuf, nullptr, NS, ND, NS, NHD, SA, SV, SV);
        epilogue_kernel<<<NS, 256, 0, stream>>>(Ybuf, osrc_w + i * 8192, osrc_b + i * 32,
                                                gn_g + i * 32, gn_b + i * 32, Xsrc);
        // ---- trg branch ----
        attn_kernel<<<NS, 256, 0, stream>>>(trg_pos, a2w1 + i * 8, a2b1 + i * 8,
                                            a2w2 + i * 64, a2b2 + i * 8, Abuf);
        vproj_kernel<<<NS, 256, 0, stream>>>(Xtrg, wtrg_w + i * 8192, wtrg_b + i * 256,
                                             proxy + i * 2048, Vbuf);
        transpose_f16_kernel<<<trV, 256, 0, stream>>>(Vbuf, VT, NS, ND);
        wmma_gemm_kernel<<<gemm_attn_blocks, 256, 0, stream>>>(
            Abuf, VT, Ybuf, nullptr, NS, ND, NS, NHD, SA, SV, SV);
        epilogue_kernel<<<NS, 256, 0, stream>>>(Ybuf, otrg_w + i * 8192, otrg_b + i * 32,
                                                gn_g + i * 32, gn_b + i * 32, Xtrg);
    }

    // final projection: out = X @ out_w + out_b   (both branches)
    // WT[n][k] = out_w[k][n]  -> K-major B operand
    cvtT_f16_kernel<<<trW, 256, 0, stream>>>(out_w, WT, ND, ND);

    float* out_f = (float*)d_out;
    cvt_f16_kernel<<<(NS * ND + 255) / 256, 256, 0, stream>>>(Xsrc, Xh, NS * ND);
    wmma_gemm_kernel<<<gemm_out_blocks, 256, 0, stream>>>(
        Xh, WT, out_f, out_b, NS, ND, ND, 1, 0, 0, 0);

    cvt_f16_kernel<<<(NS * ND + 255) / 256, 256, 0, stream>>>(Xtrg, Xh, NS * ND);
    wmma_gemm_kernel<<<gemm_out_blocks, 256, 0, stream>>>(
        Xh, WT, out_f + (size_t)NS * ND, out_b, NS, ND, ND, 1, 0, 0, 0);
}
